// UnrolledLSTMModel_56916906606613
// MI455X (gfx1250) — matrix operand order repaired and checked by reading, not compiled, not run on hardware
//
#include <hip/hip_runtime.h>

// ---------------------------------------------------------------------------
// Stacked 2-layer LSTM + dense head for MI455X (gfx1250, wave32, WMMA).
//   * f16 WMMA (v_wmma_f32_16x16x32_f16), f32 accumulate; weights packed to
//     [4H, K] f16 once per call (~29MB -> resident in 192MB L2 across steps).
//   * Per timestep: split-K GEMM. 8 waves/block each own a K-slice of
//     z = [x_t | h_{t-1}] @ [W;U], accumulate via WMMA, reduce partials in
//     LDS; wave 0 applies bias + gate nonlinearities + cell update.
//   * K-slice processed as two region loops (x@W then h@U) with constant-
//     stride per-lane pointers; the pipeline is unrolled x2 with two named
//     fragment buffer sets that swap roles (no register-rotation movs):
//     next chunk's global_load_b128s issue while current chunk's WMMAs run.
// ---------------------------------------------------------------------------

typedef __attribute__((ext_vector_type(16))) _Float16 v16h;
typedef __attribute__((ext_vector_type(8)))  _Float16 v8h;
typedef __attribute__((ext_vector_type(8)))  float    v8f;

#define LSTM_B  64
#define LSTM_T  256
#define LSTM_D  512
#define LSTM_H1 1024
#define LSTM_H2 1024
#define LSTM_O  512

__device__ __forceinline__ float sigf(float x) {
    return 1.0f / (1.0f + __expf(-x));
}

// ---- fragment assembly from per-lane pointers -------------------------------
// A matrix 16x32 f16: lane (l&15) owns row; halves 0..15 -> K = hb*16+i
// (hb = l>>4). pa already includes
// (m0 + lane&15)*stride + hb*16.  Two 16B loads per lane.
__device__ __forceinline__ v16h frag_a(const _Float16* __restrict__ pa) {
    v8h lo = *(const v8h*)(pa);
    v8h hi = *(const v8h*)(pa + 8);
    v16h a;
#pragma unroll
    for (int i = 0; i < 8; ++i) { a[i] = lo[i]; a[i + 8] = hi[i]; }
    return a;
}

// B matrix 32x16 f16 (K x N): lane (l&15) owns column; lanes 0-15 hold
// K=0..15, lanes 16-31 hold K=16..31, sequential halves. pb already includes
// (n0 + lane&15)*Kdim + hb*16 into the transposed [N,K] weights.
__device__ __forceinline__ v16h frag_b(const _Float16* __restrict__ pb) {
    v8h lo = *(const v8h*)(pb);
    v8h hi = *(const v8h*)(pb + 8);
    v16h b;
#pragma unroll
    for (int i = 0; i < 8; ++i) { b[i] = lo[i]; b[i + 8] = hi[i]; }
    return b;
}

__device__ __forceinline__ void wmma4(const v16h& a, const v16h& b0, const v16h& b1,
                                      const v16h& b2, const v16h& b3, v8f (&acc)[4]) {
    acc[0] = __builtin_amdgcn_wmma_f32_16x16x32_f16(false, a, false, b0, (short)0, acc[0], false, false);
    acc[1] = __builtin_amdgcn_wmma_f32_16x16x32_f16(false, a, false, b1, (short)0, acc[1], false, false);
    acc[2] = __builtin_amdgcn_wmma_f32_16x16x32_f16(false, a, false, b2, (short)0, acc[2], false, false);
    acc[3] = __builtin_amdgcn_wmma_f32_16x16x32_f16(false, a, false, b3, (short)0, acc[3], false, false);
}

// Pipelined GEMM over `cnt` 32-wide K chunks: 4 gate columns share one A.
// Ping-pong between two named buffer sets (unroll x2) -> zero rotation movs.
__device__ __forceinline__ void gemm_region(const _Float16* __restrict__ pa,
                                            const _Float16* __restrict__ pb0,
                                            const _Float16* __restrict__ pb1,
                                            const _Float16* __restrict__ pb2,
                                            const _Float16* __restrict__ pb3,
                                            int cnt, v8f (&acc)[4]) {
    if (cnt <= 0) return;
    v16h aA  = frag_a(pa);
    v16h bA0 = frag_b(pb0);
    v16h bA1 = frag_b(pb1);
    v16h bA2 = frag_b(pb2);
    v16h bA3 = frag_b(pb3);
    int i = 1;
    for (; i + 1 < cnt; i += 2) {
        pa += 32; pb0 += 32; pb1 += 32; pb2 += 32; pb3 += 32;
        v16h aB  = frag_a(pa);
        v16h bB0 = frag_b(pb0);
        v16h bB1 = frag_b(pb1);
        v16h bB2 = frag_b(pb2);
        v16h bB3 = frag_b(pb3);
        wmma4(aA, bA0, bA1, bA2, bA3, acc);
        pa += 32; pb0 += 32; pb1 += 32; pb2 += 32; pb3 += 32;
        aA  = frag_a(pa);
        bA0 = frag_b(pb0);
        bA1 = frag_b(pb1);
        bA2 = frag_b(pb2);
        bA3 = frag_b(pb3);
        wmma4(aB, bB0, bB1, bB2, bB3, acc);
    }
    if (i < cnt) {   // one trailing chunk
        pa += 32; pb0 += 32; pb1 += 32; pb2 += 32; pb3 += 32;
        v16h aB  = frag_a(pa);
        v16h bB0 = frag_b(pb0);
        v16h bB1 = frag_b(pb1);
        v16h bB2 = frag_b(pb2);
        v16h bB3 = frag_b(pb3);
        wmma4(aA, bA0, bA1, bA2, bA3, acc);
        wmma4(aB, bB0, bB1, bB2, bB3, acc);
    } else {
        wmma4(aA, bA0, bA1, bA2, bA3, acc);
    }
}

// ---- packing / init kernels -------------------------------------------------
__global__ void k_cvt_f32_to_f16(const float* __restrict__ src,
                                 _Float16* __restrict__ dst, int n) {
    int i = blockIdx.x * blockDim.x + threadIdx.x;
    if (i < n) dst[i] = (_Float16)src[i];
}

// src [K,N] f32 row-major -> dst [N,K] f16 (B-fragment friendly)
__global__ void k_transpose_f32_to_f16(const float* __restrict__ src,
                                       _Float16* __restrict__ dst, int K, int N) {
    int i = blockIdx.x * blockDim.x + threadIdx.x;
    if (i < K * N) {
        int k = i / N;
        int n = i - k * N;
        dst[(size_t)n * K + k] = (_Float16)src[i];
    }
}

__global__ void k_zero_f16(_Float16* __restrict__ p, int n) {
    int i = blockIdx.x * blockDim.x + threadIdx.x;
    if (i < n) p[i] = (_Float16)0.0f;
}

__global__ void k_zero_f32(float* __restrict__ p, int n) {
    int i = blockIdx.x * blockDim.x + threadIdx.x;
    if (i < n) p[i] = 0.0f;
}

// ---- one LSTM timestep (split-K) -------------------------------------------
// grid = (H/16) * (B/16) blocks of 256 threads (8 waves).
// Block computes a 16(batch) x 16(hidden) tile for all 4 gates.
// Wave w owns K-chunks [w*per, (w+1)*per) of the fused K = Din + H loop;
// the slice crosses the x->h boundary at most once -> two region loops.
__global__ void __launch_bounds__(256)
k_lstm_step(const _Float16* __restrict__ xA, int xStride,
            const _Float16* __restrict__ hA, int hStride,
            const _Float16* __restrict__ Wt, int Din,
            const _Float16* __restrict__ Ut, int H,
            const float* __restrict__ bias,
            float* __restrict__ cState,
            _Float16* __restrict__ hOut, int hOutStride) {
    __shared__ __align__(32) float red[8][4][32][8];   // [wave][gate][lane][j]

    const int tid  = threadIdx.x;
    const int w    = tid >> 5;
    const int lane = tid & 31;
    const int r    = lane & 15;
    const int hb   = lane >> 4;
    const int m0 = (blockIdx.x & 3) << 4;   // batch tile (B=64 -> 4 tiles)
    const int n0 = (blockIdx.x >> 2) << 4;  // hidden tile

    const int nChunks = (Din + H) >> 5;     // 32-wide K chunks (48 or 64)
    const int per = nChunks >> 3;           // chunks per wave (6 or 8)
    const int c0  = w * per;
    const int c1  = c0 + per;
    const int bCh = Din >> 5;               // x/h boundary chunk

    v8f acc[4];
#pragma unroll
    for (int g = 0; g < 4; ++g)
#pragma unroll
        for (int i = 0; i < 8; ++i) acc[g][i] = 0.0f;

    // region 1: x_t @ W  (chunks [c0, min(c1,bCh)))
    {
        const int e = (c1 < bCh) ? c1 : bCh;
        const int cnt = e - c0;
        if (cnt > 0) {
            const int k = c0 << 5;
            const _Float16* pa  = xA + (size_t)(m0 + r) * xStride + hb * 16 + k;
            const _Float16* pb0 = Wt + (size_t)(0 * H + n0 + r) * Din + hb * 16 + k;
            const _Float16* pb1 = Wt + (size_t)(1 * H + n0 + r) * Din + hb * 16 + k;
            const _Float16* pb2 = Wt + (size_t)(2 * H + n0 + r) * Din + hb * 16 + k;
            const _Float16* pb3 = Wt + (size_t)(3 * H + n0 + r) * Din + hb * 16 + k;
            gemm_region(pa, pb0, pb1, pb2, pb3, cnt, acc);
        }
    }
    // region 2: h_{t-1} @ U  (chunks [max(c0,bCh), c1))
    {
        const int s = (c0 > bCh) ? c0 : bCh;
        const int cnt = c1 - s;
        if (cnt > 0) {
            const int k = (s - bCh) << 5;
            const _Float16* pa  = hA + (size_t)(m0 + r) * hStride + hb * 16 + k;
            const _Float16* pb0 = Ut + (size_t)(0 * H + n0 + r) * H + hb * 16 + k;
            const _Float16* pb1 = Ut + (size_t)(1 * H + n0 + r) * H + hb * 16 + k;
            const _Float16* pb2 = Ut + (size_t)(2 * H + n0 + r) * H + hb * 16 + k;
            const _Float16* pb3 = Ut + (size_t)(3 * H + n0 + r) * H + hb * 16 + k;
            gemm_region(pa, pb0, pb1, pb2, pb3, cnt, acc);
        }
    }

    // stash partials, reduce in wave 0
#pragma unroll
    for (int g = 0; g < 4; ++g)
        *(v8f*)&red[w][g][lane][0] = acc[g];
    __syncthreads();
    if (w != 0) return;

    const int col = n0 + r;
    v8f z[4];
#pragma unroll
    for (int g = 0; g < 4; ++g) {
        v8f s = *(const v8f*)&red[0][g][lane][0];
#pragma unroll
        for (int ww = 1; ww < 8; ++ww)
            s += *(const v8f*)&red[ww][g][lane][0];
        const float bv = bias[g * H + col];
#pragma unroll
        for (int i = 0; i < 8; ++i) z[g][i] = s[i] + bv;
    }

    // gates (i,f,g,o), cell + hidden update; lane-local on C/D layout
#pragma unroll
    for (int j = 0; j < 8; ++j) {
        const int row = m0 + j + (hb << 3);
        const float zi = z[0][j], zf = z[1][j], zg = z[2][j], zo = z[3][j];
        const size_t cidx = (size_t)row * H + col;
        const float cold = cState[cidx];
        const float cn = sigf(zf) * cold + sigf(zi) * tanhf(zg);
        cState[cidx] = cn;
        const float hn = sigf(zo) * tanhf(cn);
        hOut[(size_t)row * hOutStride + col] = (_Float16)hn;
    }
}

// ---- dense head: out[64,O] = h_last[64,K] @ Wd + bd ------------------------
__global__ void __launch_bounds__(32)
k_dense(const _Float16* __restrict__ A, int aStride,
        const _Float16* __restrict__ Bt, int K,
        const float* __restrict__ bias,
        float* __restrict__ out, int N) {
    const int lane = threadIdx.x & 31;
    const int r    = lane & 15;
    const int hb   = lane >> 4;
    const int m0 = (blockIdx.x & 3) << 4;
    const int n0 = (blockIdx.x >> 2) << 4;
    const int col = n0 + r;
    v8f acc;
    const float bv = bias[col];
#pragma unroll
    for (int i = 0; i < 8; ++i) acc[i] = bv;

    const _Float16* pa = A  + (size_t)(m0 + r) * aStride + hb * 16;
    const _Float16* pb = Bt + (size_t)(n0 + r) * K + hb * 16;
    // ping-pong pipelined K loop (K multiple of 64 here)
    v16h aA = frag_a(pa);
    v16h bA = frag_b(pb);
    int k0 = 32;
    for (; k0 + 32 < K; k0 += 64) {
        pa += 32; pb += 32;
        v16h aB = frag_a(pa);
        v16h bB = frag_b(pb);
        acc = __builtin_amdgcn_wmma_f32_16x16x32_f16(false, aA, false, bA, (short)0, acc, false, false);
        pa += 32; pb += 32;
        aA = frag_a(pa);
        bA = frag_b(pb);
        acc = __builtin_amdgcn_wmma_f32_16x16x32_f16(false, aB, false, bB, (short)0, acc, false, false);
    }
    if (k0 < K) {
        pa += 32; pb += 32;
        v16h aB = frag_a(pa);
        v16h bB = frag_b(pb);
        acc = __builtin_amdgcn_wmma_f32_16x16x32_f16(false, aA, false, bA, (short)0, acc, false, false);
        acc = __builtin_amdgcn_wmma_f32_16x16x32_f16(false, aB, false, bB, (short)0, acc, false, false);
    } else {
        acc = __builtin_amdgcn_wmma_f32_16x16x32_f16(false, aA, false, bA, (short)0, acc, false, false);
    }

#pragma unroll
    for (int j = 0; j < 8; ++j) {
        const int row = m0 + j + (hb << 3);
        out[(size_t)row * N + col] = acc[j];
    }
}

// ---------------------------------------------------------------------------
extern "C" void kernel_launch(void* const* d_in, const int* in_sizes, int n_in,
                              void* d_out, int out_size, void* d_ws, size_t ws_size,
                              hipStream_t stream) {
    (void)in_sizes; (void)n_in; (void)out_size; (void)ws_size;

    const float* x  = (const float*)d_in[0];
    const float* W1 = (const float*)d_in[1];
    const float* U1 = (const float*)d_in[2];
    const float* b1 = (const float*)d_in[3];
    const float* W2 = (const float*)d_in[4];
    const float* U2 = (const float*)d_in[5];
    const float* b2 = (const float*)d_in[6];
    const float* Wd = (const float*)d_in[7];
    const float* bd = (const float*)d_in[8];
    float* out = (float*)d_out;

    const int B = LSTM_B, T = LSTM_T, D = LSTM_D;
    const int H1 = LSTM_H1, H2 = LSTM_H2, O = LSTM_O;

    // carve workspace (all offsets 256B-aligned)
    size_t off = 0;
    auto carve = [&](size_t elems, size_t esize) -> char* {
        char* p = (char*)d_ws + off;
        off += (elems * esize + 255) & ~(size_t)255;
        return p;
    };
    _Float16* xh   = (_Float16*)carve((size_t)B * T * D, 2);      // x in f16
    _Float16* W1t  = (_Float16*)carve((size_t)D * 4 * H1, 2);     // [4H1, D]
    _Float16* U1t  = (_Float16*)carve((size_t)H1 * 4 * H1, 2);    // [4H1, H1]
    _Float16* W2t  = (_Float16*)carve((size_t)H1 * 4 * H2, 2);    // [4H2, H1]
    _Float16* U2t  = (_Float16*)carve((size_t)H2 * 4 * H2, 2);    // [4H2, H2]
    _Float16* Wdt  = (_Float16*)carve((size_t)H2 * O, 2);         // [O, H2]
    _Float16* seq1 = (_Float16*)carve((size_t)B * T * H1, 2);     // layer-1 h seq [B,T,H1]
    _Float16* h2a  = (_Float16*)carve((size_t)B * H2, 2);         // layer-2 h ping
    _Float16* h2b  = (_Float16*)carve((size_t)B * H2, 2);         // layer-2 h pong
    _Float16* hz   = (_Float16*)carve((size_t)B * H1, 2);         // zero h0
    float*    c1   = (float*)carve((size_t)B * H1, 4);            // layer-1 c state
    float*    c2   = (float*)carve((size_t)B * H2, 4);            // layer-2 c state

    auto g1 = [](size_t n) { return dim3((unsigned)((n + 255) / 256)); };

    // state init (deterministic each call)
    k_zero_f16<<<g1((size_t)B * H1), 256, 0, stream>>>(hz, B * H1);
    k_zero_f32<<<g1((size_t)B * H1), 256, 0, stream>>>(c1, B * H1);
    k_zero_f32<<<g1((size_t)B * H2), 256, 0, stream>>>(c2, B * H2);

    // pack inputs/weights to f16 (weights transposed for B-fragments)
    k_cvt_f32_to_f16<<<g1((size_t)B * T * D), 256, 0, stream>>>(x, xh, B * T * D);
    k_transpose_f32_to_f16<<<g1((size_t)D * 4 * H1), 256, 0, stream>>>(W1, W1t, D, 4 * H1);
    k_transpose_f32_to_f16<<<g1((size_t)H1 * 4 * H1), 256, 0, stream>>>(U1, U1t, H1, 4 * H1);
    k_transpose_f32_to_f16<<<g1((size_t)H1 * 4 * H2), 256, 0, stream>>>(W2, W2t, H1, 4 * H2);
    k_transpose_f32_to_f16<<<g1((size_t)H2 * 4 * H2), 256, 0, stream>>>(U2, U2t, H2, 4 * H2);
    k_transpose_f32_to_f16<<<g1((size_t)H2 * O), 256, 0, stream>>>(Wd, Wdt, H2, O);

    // ---- layer 1: h_t written into seq1[:, t, :]; h_{t-1} read from seq1 ----
    const dim3 stepGrid1((H1 / 16) * (B / 16));
    for (int t = 0; t < T; ++t) {
        const _Float16* hA = (t == 0) ? hz : (seq1 + (size_t)(t - 1) * H1);
        const int hStride = (t == 0) ? H1 : T * H1;
        k_lstm_step<<<stepGrid1, 256, 0, stream>>>(
            xh + (size_t)t * D, T * D,
            hA, hStride,
            W1t, D, U1t, H1, b1, c1,
            seq1 + (size_t)t * H1, T * H1);
    }

    // ---- layer 2: consumes seq1, ping-pongs h; only h_last needed ----------
    const dim3 stepGrid2((H2 / 16) * (B / 16));
    for (int t = 0; t < T; ++t) {
        const _Float16* hA = (t == 0) ? hz : ((t & 1) ? h2a : h2b);
        _Float16* hO = (t & 1) ? h2b : h2a;
        k_lstm_step<<<stepGrid2, 256, 0, stream>>>(
            seq1 + (size_t)t * H1, T * H1,
            hA, H2,
            W2t, H1, U2t, H2, b2, c2,
            hO, H2);
    }

    // ---- dense head ---------------------------------------------------------
    const _Float16* hLast = ((T - 1) & 1) ? h2b : h2a;  // T=256 -> last write in h2b
    k_dense<<<dim3((O / 16) * (B / 16)), 32, 0, stream>>>(
        hLast, H2, Wdt, H2, bd, out, O);
}
